// MultiHeadAttention_15908558864750
// MI455X (gfx1250) — matrix operand order repaired and checked by reading, not compiled, not run on hardware
//
#include <hip/hip_runtime.h>
#include <hip/hip_bf16.h>

// ---------------------------------------------------------------------------
// Types for CDNA5 WMMA (wave32): A/B = 16 bf16 per lane, C/D = 8 f32 per lane.
// ---------------------------------------------------------------------------
typedef __bf16 bf16_t;
typedef __bf16 v16bf __attribute__((ext_vector_type(16)));
typedef __bf16 v8bf  __attribute__((ext_vector_type(8)));
typedef float  v8f   __attribute__((ext_vector_type(8)));

#define WMMA_BF16(A_, B_, C_) \
  __builtin_amdgcn_wmma_f32_16x16x32_bf16(false, (A_), false, (B_), (short)0, (C_), false, false)

static constexpr int D_MODEL = 1024;
static constexpr int NHEADS  = 16;
static constexpr int HDIM    = 64;
static constexpr int NTOK    = 2048;
static constexpr int BATCH   = 2;
static constexpr int MROWS   = BATCH * NTOK;   // 4096

__device__ __forceinline__ v8f zero_v8f() {
  v8f z = {0.f, 0.f, 0.f, 0.f, 0.f, 0.f, 0.f, 0.f};
  return z;
}

// B-operand fragment (K x 16): lane L holds column N = L&15, 16 contiguous K
// values starting at K = 16*(L>>4).  One 32-byte contiguous load per lane.
__device__ __forceinline__ v16bf load_b_frag(const bf16_t* p) {
  return *(const v16bf*)p;
}

// A-operand fragment (16 x 32): lane L holds row M = L&15; elements 0..15 are
// K = 16*(L>>4)..+15 (contiguous, matching load_b_frag).
// Two 16-byte contiguous loads per lane.
__device__ __forceinline__ v16bf load_a_frag(const bf16_t* row, int kbase, int hi) {
  v8bf lo  = *(const v8bf*)(row + kbase + 16 * hi);
  v8bf hi8 = *(const v8bf*)(row + kbase + 16 * hi + 8);
  return __builtin_shufflevector(lo, hi8, 0, 1, 2, 3, 4, 5, 6, 7,
                                          8, 9, 10, 11, 12, 13, 14, 15);
}

// ---------------------------------------------------------------------------
// fp32 -> bf16 convert
// ---------------------------------------------------------------------------
__global__ void cvt_f32_bf16(const float* __restrict__ in,
                             bf16_t* __restrict__ out, int n) {
  int i = blockIdx.x * blockDim.x + threadIdx.x;
  if (i < n) out[i] = (bf16_t)in[i];
}

// ---------------------------------------------------------------------------
// NT GEMM:  out[m,n] = sum_k A[m,k] * W[n,k] + bias[n]
// A: [4096,1024] bf16 row-major, W: [1024,1024] bf16 row-major.
// Block = 256 threads (8 waves), tile 128x128, each wave 32x64 (2x4 WMMA frags).
// A/B tiles staged in double-buffered LDS; next k-step's global loads are
// issued right after the barrier so they overlap the current 8 WMMAs.
// mode 0: bf16 out, head-split   [B*H, NTOK, HDIM]   (Q, K)
// mode 1: bf16 out, head-split-T [B*H, HDIM, NTOK]   (V, for NT P@V)
// mode 2: f32  out, row-major    [M, 1024]           (final projection)
// ---------------------------------------------------------------------------
__global__ __launch_bounds__(256) void gemm_nt_bf16(
    const bf16_t* __restrict__ A,
    const bf16_t* __restrict__ W,
    const float*  __restrict__ bias,
    void* __restrict__ outp, int mode)
{
  constexpr int K  = D_MODEL;
  constexpr int NK = K / 32;                 // 32 k-steps

  __shared__ bf16_t As[2][128 * 32];         // [tile row][k]   8KB each
  __shared__ bf16_t Bs[2][128 * 32];         // [tile col][k]   8KB each

  const int tid  = threadIdx.x;
  const int lane = tid & 31;
  const int wave = tid >> 5;
  const int hi = lane >> 4;
  const int lx = lane & 15;
  const int wr = wave & 3;   // 4 waves along M (32 rows each)
  const int wc = wave >> 2;  // 2 waves along N (64 cols each)
  const int tileM = blockIdx.y * 128;
  const int tileN = blockIdx.x * 128;

  // Staging: thread t copies one 32-byte half-row of A and of B per k-step.
  const int crow = tid >> 1;               // 0..127
  const int koff = (tid & 1) * 16;         // 0 or 16
  const bf16_t* gA = A + (size_t)(tileM + crow) * K + koff;
  const bf16_t* gB = W + (size_t)(tileN + crow) * K + koff;
  const int lidx = crow * 32 + koff;

  v8f acc[2][4];
#pragma unroll
  for (int mi = 0; mi < 2; ++mi)
#pragma unroll
    for (int ni = 0; ni < 4; ++ni)
      acc[mi][ni] = zero_v8f();

  // preload k-step 0
  v16bf ra = *(const v16bf*)gA;
  v16bf rb = *(const v16bf*)gB;

  for (int i = 0; i < NK; ++i) {
    const int buf = i & 1;
    *(v16bf*)(&As[buf][lidx]) = ra;
    *(v16bf*)(&Bs[buf][lidx]) = rb;
    __syncthreads();
    if (i + 1 < NK) {                      // in-flight during the WMMAs below
      ra = *(const v16bf*)(gA + (i + 1) * 32);
      rb = *(const v16bf*)(gB + (i + 1) * 32);
    }
    const bf16_t* as = &As[buf][0];
    const bf16_t* bs = &Bs[buf][0];
    v16bf a0 = load_a_frag(as + (wr * 32 + lx) * 32, 0, hi);
    v16bf a1 = load_a_frag(as + (wr * 32 + 16 + lx) * 32, 0, hi);
    v16bf b[4];
#pragma unroll
    for (int ni = 0; ni < 4; ++ni)
      b[ni] = load_b_frag(bs + (wc * 64 + ni * 16 + lx) * 32 + 16 * hi);
#pragma unroll
    for (int ni = 0; ni < 4; ++ni) {
      acc[0][ni] = WMMA_BF16(a0, b[ni], acc[0][ni]);
      acc[1][ni] = WMMA_BF16(a1, b[ni], acc[1][ni]);
    }
    // single barrier per iteration is safe with double buffering: the next
    // write to this buffer is two iterations away, past the next barrier.
  }

  float bsel[4];
#pragma unroll
  for (int ni = 0; ni < 4; ++ni) bsel[ni] = bias[tileN + wc * 64 + ni * 16 + lx];

#pragma unroll
  for (int mi = 0; mi < 2; ++mi) {
#pragma unroll
    for (int ni = 0; ni < 4; ++ni) {
      const int n = tileN + wc * 64 + ni * 16 + lx;
#pragma unroll
      for (int r = 0; r < 8; ++r) {
        float v = acc[mi][ni][r] + bsel[ni];
        const int m = tileM + wr * 32 + mi * 16 + r + 8 * hi;  // C: M = r + 8*(L>>4)
        if (mode == 2) {
          ((float*)outp)[(size_t)m * D_MODEL + n] = v;
        } else {
          const int token = m & (NTOK - 1);
          const int bb = m >> 11;
          const int h = n >> 6;
          const int d = n & (HDIM - 1);
          size_t idx;
          if (mode == 0)
            idx = ((size_t)(bb * NHEADS + h) * NTOK + token) * HDIM + d;
          else
            idx = ((size_t)(bb * NHEADS + h) * HDIM + d) * NTOK + token;
          ((bf16_t*)outp)[idx] = (bf16_t)v;
        }
      }
    }
  }
}

// ---------------------------------------------------------------------------
// Flash-style attention (no score materialization, no 1/sqrt(dh) scaling,
// faithful to the reference).  Block = 256 threads = 8 waves on one (b,h);
// each wave owns 16 query rows.  Keys stream in blocks of 32: the 32x64 K
// block and 64x32 V^T block are staged once per workgroup into double-
// buffered LDS (one 16B global load per thread per block each), then all 8
// waves build WMMA fragments from LDS.  8 WMMAs per block per wave.
// ---------------------------------------------------------------------------
__global__ __launch_bounds__(256) void attn_fwd(
    const bf16_t* __restrict__ Qh,  // [B*H, NTOK, HDIM]
    const bf16_t* __restrict__ Kh,  // [B*H, NTOK, HDIM]
    const bf16_t* __restrict__ Vt,  // [B*H, HDIM, NTOK]
    bf16_t* __restrict__ Ao)        // [B, NTOK, D_MODEL]
{
  __shared__ bf16_t Kl[2][32 * 64];   // [key j][d]    4KB each
  __shared__ bf16_t Vl[2][64 * 32];   // [d][key j]    4KB each
  __shared__ bf16_t Pl[8][16 * 32];   // per-wave P transpose buffer

  const int tid  = threadIdx.x;
  const int lane = tid & 31;
  const int wave = tid >> 5;
  const int hi = lane >> 4;
  const int lx = lane & 15;
  const int bh = blockIdx.y;                  // 0..31  (b*16 + h)
  const int q0 = blockIdx.x * 128 + wave * 16;

  const bf16_t* Qb = Qh + (size_t)bh * NTOK * HDIM;
  const bf16_t* Kb = Kh + (size_t)bh * NTOK * HDIM;
  const bf16_t* Vb = Vt + (size_t)bh * HDIM * NTOK;

  // Q fragments for the two dh K-steps (d 0..31, d 32..63); loaded once.
  const bf16_t* qrow = Qb + (size_t)(q0 + lx) * HDIM;
  v16bf aq0 = load_a_frag(qrow, 0, hi);
  v16bf aq1 = load_a_frag(qrow, 32, hi);

  // Staging addresses: K block is 4KB contiguous; V block is 64 rows x 64B.
  const int kofs = tid * 8;                    // 8 bf16 = 16B per thread
  const int vrow = tid >> 2;                   // d = 0..63
  const int vofs = (tid & 3) * 8;              // j offset within block
  const bf16_t* gK = Kb + kofs;                // + j0*HDIM per block
  const bf16_t* gV = Vb + (size_t)vrow * NTOK + vofs;  // + j0 per block

  float m_i[8], l_i[8];
#pragma unroll
  for (int r = 0; r < 8; ++r) { m_i[r] = -3.0e38f; l_i[r] = 0.f; }

  v8f o[4];
#pragma unroll
  for (int df = 0; df < 4; ++df) o[df] = zero_v8f();

  bf16_t* pbuf = &Pl[wave][0];   // 16x32 bf16, row stride 32

  // preload key-block 0
  v8bf kreg = *(const v8bf*)gK;
  v8bf vreg = *(const v8bf*)gV;

  for (int jb = 0; jb < NTOK / 32; ++jb) {
    const int buf = jb & 1;
    *(v8bf*)(&Kl[buf][kofs]) = kreg;
    *(v8bf*)(&Vl[buf][vrow * 32 + vofs]) = vreg;
    __syncthreads();
    if (jb + 1 < NTOK / 32) {               // overlap with this block's math
      kreg = *(const v8bf*)(gK + (jb + 1) * 32 * HDIM);
      vreg = *(const v8bf*)(gV + (jb + 1) * 32);
    }

    // ---- S = Q K^T for 16 queries x 32 keys (two 16-col fragments) ----
    const bf16_t* kl = &Kl[buf][0];
    v8f s0 = zero_v8f();
    v8f s1 = zero_v8f();
    {
      const bf16_t* kr0 = kl + (lx)*HDIM + 16 * hi;
      const bf16_t* kr1 = kl + (16 + lx) * HDIM + 16 * hi;
      v16bf b00 = load_b_frag(kr0);        // keys 0..15,  d 0..31
      v16bf b01 = load_b_frag(kr0 + 32);   // keys 0..15,  d 32..63
      v16bf b10 = load_b_frag(kr1);        // keys 16..31, d 0..31
      v16bf b11 = load_b_frag(kr1 + 32);
      s0 = WMMA_BF16(aq0, b00, s0);
      s0 = WMMA_BF16(aq1, b01, s0);
      s1 = WMMA_BF16(aq0, b10, s1);
      s1 = WMMA_BF16(aq1, b11, s1);
    }

    // ---- online softmax over this key block (row = r + 8*hi) ----
    float scale[8];
#pragma unroll
    for (int r = 0; r < 8; ++r) {
      float v0 = s0[r], v1 = s1[r];
      float mx = fmaxf(v0, v1);
#pragma unroll
      for (int off = 1; off < 16; off <<= 1)
        mx = fmaxf(mx, __shfl_xor(mx, off, 32));   // reduce within 16-lane half
      float mnew = fmaxf(m_i[r], mx);
      float sc = __expf(m_i[r] - mnew);
      float p0 = __expf(v0 - mnew);
      float p1 = __expf(v1 - mnew);
      float rs = p0 + p1;
#pragma unroll
      for (int off = 1; off < 16; off <<= 1)
        rs += __shfl_xor(rs, off, 32);
      l_i[r] = l_i[r] * sc + rs;
      m_i[r] = mnew;
      scale[r] = sc;
      // stage P (C-layout -> row-major 16x32) into per-wave LDS
      pbuf[(r + 8 * hi) * 32 + lx] = (bf16_t)p0;
      pbuf[(r + 8 * hi) * 32 + lx + 16] = (bf16_t)p1;
    }
    asm volatile("s_wait_dscnt 0x0" ::: "memory");  // LDS RAW within wave

    // rescale running O by exp(m_old - m_new)
#pragma unroll
    for (int df = 0; df < 4; ++df)
#pragma unroll
      for (int r = 0; r < 8; ++r)
        o[df][r] *= scale[r];

    // ---- O += P @ V  (P 16x32 from LDS in A-layout, V^T rows contiguous) ----
    v16bf ap = load_a_frag(pbuf + lx * 32, 0, hi);
    const bf16_t* vl = &Vl[buf][0];
#pragma unroll
    for (int df = 0; df < 4; ++df) {
      v16bf bv = load_b_frag(vl + (df * 16 + lx) * 32 + 16 * hi);
      o[df] = WMMA_BF16(ap, bv, o[df]);
    }
    // single barrier per iteration (next overwrite of this buffer is two
    // iterations away, past the next barrier)
  }

  // ---- finalize: divide by row sums, scatter to [B, NTOK, H*HDIM] bf16 ----
  float inv[8];
#pragma unroll
  for (int r = 0; r < 8; ++r) inv[r] = 1.0f / l_i[r];
  const int b = bh >> 4, h = bh & 15;
#pragma unroll
  for (int df = 0; df < 4; ++df) {
#pragma unroll
    for (int r = 0; r < 8; ++r) {
      const int token = q0 + r + 8 * hi;
      size_t idx = ((size_t)b * NTOK + token) * D_MODEL + h * HDIM + df * 16 + lx;
      Ao[idx] = (bf16_t)(o[df][r] * inv[r]);
    }
  }
}

// ---------------------------------------------------------------------------
// Host launcher
// ---------------------------------------------------------------------------
extern "C" void kernel_launch(void* const* d_in, const int* in_sizes, int n_in,
                              void* d_out, int out_size, void* d_ws, size_t ws_size,
                              hipStream_t stream) {
  (void)in_sizes; (void)n_in; (void)out_size; (void)ws_size;

  const float* q_in = (const float*)d_in[0];
  const float* k_in = (const float*)d_in[1];
  const float* v_in = (const float*)d_in[2];
  const float* Wq = (const float*)d_in[3];
  const float* bq = (const float*)d_in[4];
  const float* Wk = (const float*)d_in[5];
  const float* bk = (const float*)d_in[6];
  const float* Wv = (const float*)d_in[7];
  const float* bv = (const float*)d_in[8];
  const float* Wo = (const float*)d_in[9];
  const float* bo = (const float*)d_in[10];

  const size_t ACT = (size_t)BATCH * NTOK * D_MODEL;  // 4,194,304 elems
  const size_t WEL = (size_t)D_MODEL * D_MODEL;       // 1,048,576 elems

  char* w = (char*)d_ws;
  bf16_t* qx  = (bf16_t*)w; w += ACT * 2;
  bf16_t* kx  = (bf16_t*)w; w += ACT * 2;
  bf16_t* vx  = (bf16_t*)w; w += ACT * 2;
  bf16_t* wqb = (bf16_t*)w; w += WEL * 2;
  bf16_t* wkb = (bf16_t*)w; w += WEL * 2;
  bf16_t* wvb = (bf16_t*)w; w += WEL * 2;
  bf16_t* wob = (bf16_t*)w; w += WEL * 2;
  bf16_t* Qh  = (bf16_t*)w; w += ACT * 2;  // [B*H, NTOK, HDIM]
  bf16_t* Kh  = (bf16_t*)w; w += ACT * 2;
  bf16_t* Vtb = (bf16_t*)w; w += ACT * 2;  // [B*H, HDIM, NTOK]
  bf16_t* Ao  = (bf16_t*)w; w += ACT * 2;  // [B, NTOK, D_MODEL]

  {
    const int na = (int)ACT, nw = (int)WEL;
    cvt_f32_bf16<<<(na + 255) / 256, 256, 0, stream>>>(q_in, qx, na);
    cvt_f32_bf16<<<(na + 255) / 256, 256, 0, stream>>>(k_in, kx, na);
    cvt_f32_bf16<<<(na + 255) / 256, 256, 0, stream>>>(v_in, vx, na);
    cvt_f32_bf16<<<(nw + 255) / 256, 256, 0, stream>>>(Wq, wqb, nw);
    cvt_f32_bf16<<<(nw + 255) / 256, 256, 0, stream>>>(Wk, wkb, nw);
    cvt_f32_bf16<<<(nw + 255) / 256, 256, 0, stream>>>(Wv, wvb, nw);
    cvt_f32_bf16<<<(nw + 255) / 256, 256, 0, stream>>>(Wo, wob, nw);
  }

  dim3 gg(D_MODEL / 128, MROWS / 128);  // (8, 32) tiles of 128x128
  gemm_nt_bf16<<<gg, 256, 0, stream>>>(qx, wqb, bq, Qh, 0);
  gemm_nt_bf16<<<gg, 256, 0, stream>>>(kx, wkb, bk, Kh, 0);
  gemm_nt_bf16<<<gg, 256, 0, stream>>>(vx, wvb, bv, Vtb, 1);

  attn_fwd<<<dim3(NTOK / 128, BATCH * NHEADS), 256, 0, stream>>>(Qh, Kh, Vtb, Ao);

  gemm_nt_bf16<<<gg, 256, 0, stream>>>(Ao, wob, bo, d_out, 2);
}
